// CaptioningTransformer_62380105008153
// MI455X (gfx1250) — compile-verified, matrix operand order checked by reading, not run on hardware
//
#include <hip/hip_runtime.h>

// ---------------- problem constants ----------------
constexpr int N_  = 8;
constexpr int T_  = 512;
constexpr int D_  = 1024;
constexpr int W_  = 512;
constexpr int H_  = 4;
constexpr int HD_ = W_ / H_;     // 128
constexpr int V_  = 32000;
constexpr int L_  = 2;
constexpr int FF_ = 2048;
constexpr int NT_ = N_ * T_;     // 4096

// ---------------- WMMA types ----------------
typedef __attribute__((ext_vector_type(16))) __bf16 bf16x16;
typedef __attribute__((ext_vector_type(8)))  float  floatx8;

union bf16frag {
    bf16x16 v;
    unsigned int u[8];
};

__device__ __forceinline__ unsigned short f2bf(float f) {
    unsigned u = __float_as_uint(f);
    unsigned r = u + 0x7FFFu + ((u >> 16) & 1u);   // RNE to bf16
    return (unsigned short)(r >> 16);
}
__device__ __forceinline__ unsigned int pack_bf(float lo, float hi) {
    return (unsigned int)f2bf(lo) | ((unsigned int)f2bf(hi) << 16);
}
__device__ __forceinline__ floatx8 wmma_bf16(const bf16frag& a, const bf16frag& b, floatx8 c) {
    return __builtin_amdgcn_wmma_f32_16x16x32_bf16(false, a.v, false, b.v,
                                                   (short)0, c, false, false);
}

// ------------- CDNA5 async global->LDS DMA (ASYNCcnt-tracked) -------------
// LDS aperture low 32 bits == LDS byte offset (ISA 10.2)
__device__ __forceinline__ unsigned lds_off(const void* p) {
    return (unsigned)(unsigned long long)(size_t)p;
}
__device__ __forceinline__ void async_b128(void* lds, const void* g) {
    asm volatile("global_load_async_to_lds_b128 %0, %1, off"
                 :: "v"(lds_off(lds)), "v"(g) : "memory");
}
__device__ __forceinline__ void wait_async_0() {
    asm volatile("s_wait_asynccnt 0x0" ::: "memory");
}
__device__ __forceinline__ void wait_async_8() {   // one 8-op tile may stay in flight
    asm volatile("s_wait_asynccnt 0x8" ::: "memory");
}

// ---------------- embedding + positional encoding ----------------
__global__ void embed_kernel(const int* __restrict__ captions,
                             const float* __restrict__ emb,
                             const float* __restrict__ pe,
                             float* __restrict__ x) {
    int idx = blockIdx.x * 256 + threadIdx.x;
    if (idx >= NT_ * W_) return;
    int row = idx / W_, c = idx % W_;
    int t = row % T_;
    int cap = captions[row];
    x[idx] = emb[(size_t)cap * W_ + c] + pe[(size_t)t * W_ + c];
}

// ---------------- tiny GEMM (M = 8 paths: vis proj, cross-attn) ----------------
__global__ void small_mm(const float* __restrict__ in, const float* __restrict__ w,
                         const float* __restrict__ b, float* __restrict__ out,
                         int M, int K, int Nn) {
    int idx = blockIdx.x * blockDim.x + threadIdx.x;
    if (idx >= M * Nn) return;
    int i = idx / Nn, j = idx % Nn;
    float acc = b[j];
    for (int kk = 0; kk < K; ++kk)
        acc += in[(size_t)i * K + kk] * w[(size_t)kk * Nn + j];
    out[idx] = acc;
}

// ---------------- main WMMA GEMM: C = A(MxK) @ B(KxN) + bias, opt ReLU -------
// block = 256 threads (8 waves), tile 128x128, K step 32.
// Pipeline: async b128 DMA raw f32 tiles (double-buffered, 8 ops/lane/tile,
// s_wait_asynccnt 8 fences the older tile) -> one-touch f32->bf16 convert pass
// into compact dword layouts -> inner loop is pure ds_load_b128 + v_wmma.
__global__ __launch_bounds__(256) void gemm_bf16(
        const float* __restrict__ A, const float* __restrict__ B,
        const float* __restrict__ bias, float* __restrict__ C,
        int K, int Nn, int relu) {
    __shared__ __align__(16) float    rawA[2][128 * 32];  // [m][k] f32 (DMA dest)
    __shared__ __align__(16) float    rawB[2][32 * 128];  // [k][n] f32 (DMA dest)
    __shared__ __align__(16) unsigned sA16[128 * 16];     // [m][k/2] bf16 pairs
    __shared__ __align__(16) unsigned sB16[128 * 16];     // [n][k/2] bf16 pairs

    int tid  = threadIdx.x;
    int wid  = tid >> 5;
    int lane = tid & 31;
    int l15  = lane & 15;
    int hb   = lane >> 4;
    int row0 = blockIdx.y * 128;
    int col0 = blockIdx.x * 128;
    int wm   = (wid & 3) * 32;      // wave M offset in tile
    int wn   = (wid >> 2) * 64;     // wave N offset in tile

    // dword index of bf16-pair j within a 16-dword fragment row
    int djof[8];
#pragma unroll
    for (int j = 0; j < 8; ++j)
        djof[j] = (j & 3) + ((j & 4) ? 8 : 0) + hb * 4;

    floatx8 acc[2][4];
#pragma unroll
    for (int t = 0; t < 2; ++t)
#pragma unroll
        for (int u = 0; u < 4; ++u)
            acc[t][u] = (floatx8){0.f,0.f,0.f,0.f,0.f,0.f,0.f,0.f};

    // async-stage one raw 128x32 A tile + 32x128 B tile (8 b128 ops per lane)
    auto stage = [&](int buf, int k0) {
#pragma unroll
        for (int i = tid; i < 1024; i += 256) {           // A: 1024 16B chunks
            int r = i >> 3, cg = (i & 7) << 2;
            async_b128(&rawA[buf][r * 32 + cg],
                       &A[(size_t)(row0 + r) * K + k0 + cg]);
        }
#pragma unroll
        for (int i = tid; i < 1024; i += 256) {           // B: 1024 16B chunks
            int r = i >> 5, cg = (i & 31) << 2;           // r = k, cg = n
            async_b128(&rawB[buf][r * 128 + cg],
                       &B[(size_t)(k0 + r) * Nn + col0 + cg]);
        }
    };

    int nk = K >> 5;
    stage(0, 0);
    for (int it = 0; it < nk; ++it) {
        int cur = it & 1;
        bool pre = (it + 1 < nk);
        if (pre) stage(cur ^ 1, (it + 1) << 5);     // overlap next tile DMA
        if (pre) wait_async_8(); else wait_async_0();
        __syncthreads();                            // raw[cur] visible block-wide

        // one-touch convert: f32 -> packed bf16 (A copy, B transpose)
#pragma unroll
        for (int i = tid; i < 2048; i += 256) {
            int m = i >> 4, kp = i & 15;
            float2 f = *(const float2*)&rawA[cur][m * 32 + kp * 2];
            sA16[m * 16 + kp] = pack_bf(f.x, f.y);
        }
#pragma unroll
        for (int i = tid; i < 2048; i += 256) {
            int n = i >> 4, kp = i & 15;
            float f0 = rawB[cur][(kp * 2)     * 128 + n];
            float f1 = rawB[cur][(kp * 2 + 1) * 128 + n];
            sB16[n * 16 + kp] = pack_bf(f0, f1);
        }
        __syncthreads();                            // bf16 tiles ready

        bf16frag af[2], bfr[4];
#pragma unroll
        for (int t = 0; t < 2; ++t) {
            int m = wm + t * 16 + l15;
#pragma unroll
            for (int j = 0; j < 8; ++j)
                af[t].u[j] = sA16[m * 16 + djof[j]];
        }
#pragma unroll
        for (int u = 0; u < 4; ++u) {
            int nn = wn + u * 16 + l15;
#pragma unroll
            for (int j = 0; j < 8; ++j)
                bfr[u].u[j] = sB16[nn * 16 + djof[j]];
        }
#pragma unroll
        for (int t = 0; t < 2; ++t)
#pragma unroll
            for (int u = 0; u < 4; ++u)
                acc[t][u] = wmma_bf16(af[t], bfr[u], acc[t][u]);
        __syncthreads();                            // done reading bf16 tiles
    }

    // epilogue: C-layout VGPR r -> row r + 8*hb, col = lane&15
#pragma unroll
    for (int t = 0; t < 2; ++t)
#pragma unroll
        for (int u = 0; u < 4; ++u)
#pragma unroll
            for (int r = 0; r < 8; ++r) {
                int row = row0 + wm + t * 16 + r + 8 * hb;
                int col = col0 + wn + u * 16 + l15;
                float vv = acc[t][u][r] + bias[col];
                if (relu) vv = fmaxf(vv, 0.f);
                C[(size_t)row * Nn + col] = vv;
            }
}

// ---------------- causal flash attention, one wave per (n,h,16 rows) --------
__global__ __launch_bounds__(32) void attn_causal(
        const float* __restrict__ q, const float* __restrict__ k,
        const float* __restrict__ v, float* __restrict__ y) {
    __shared__ __align__(16) float sK[32 * 128];     // [key][feat]
    __shared__ __align__(16) float sV[32 * 128];     // [key][feat]
    __shared__ unsigned short sP[16 * 32];           // P tile bf16 [row][key]

    int lane = threadIdx.x;
    int l15 = lane & 15, hb = lane >> 4;
    int m0 = blockIdx.x * 16;
    int h  = blockIdx.y;
    int n  = blockIdx.z;
    size_t base = ((size_t)n * T_) * W_ + (size_t)h * HD_;

    int kof[8];
#pragma unroll
    for (int j = 0; j < 8; ++j)
        kof[j] = ((j & 4) ? 16 : 0) + (j & 3) * 2 + hb * 8;

    // load Q (16 x 128) once as 4 A-fragments (K=32 each)
    bf16frag qa[4];
    {
        size_t qrow = base + (size_t)(m0 + l15) * W_;
#pragma unroll
        for (int kk = 0; kk < 4; ++kk)
#pragma unroll
            for (int j = 0; j < 8; ++j) {
                float2 f = *(const float2*)&q[qrow + kk * 32 + kof[j]];
                qa[kk].u[j] = pack_bf(f.x, f.y);
            }
    }

    floatx8 o[8];
#pragma unroll
    for (int cc = 0; cc < 8; ++cc)
        o[cc] = (floatx8){0.f,0.f,0.f,0.f,0.f,0.f,0.f,0.f};
    float mi[8], li[8];
#pragma unroll
    for (int r = 0; r < 8; ++r) { mi[r] = -3.0e38f; li[r] = 0.f; }

    const float scale = 0.08838834764831845f;    // 1/sqrt(128)
    int kend = m0 + 16;

    for (int t0 = 0; t0 < kend; t0 += 32) {
        // async-stage K and V tiles (32 keys x 128 feats) into LDS
#pragma unroll
        for (int ii = 0; ii < 32; ++ii) {
            async_b128(&sK[ii * 128 + lane * 4],
                       &k[base + (size_t)(t0 + ii) * W_ + lane * 4]);
            async_b128(&sV[ii * 128 + lane * 4],
                       &v[base + (size_t)(t0 + ii) * W_ + lane * 4]);
        }
        wait_async_0();
        __syncthreads();

        floatx8 s0 = (floatx8){0.f,0.f,0.f,0.f,0.f,0.f,0.f,0.f};
        floatx8 s1 = s0;
#pragma unroll
        for (int kk = 0; kk < 4; ++kk) {
            bf16frag kb0, kb1;
#pragma unroll
            for (int j = 0; j < 8; ++j) {
                float2 f = *(const float2*)&sK[l15 * 128 + kk * 32 + kof[j]];
                kb0.u[j] = pack_bf(f.x, f.y);
                f = *(const float2*)&sK[(16 + l15) * 128 + kk * 32 + kof[j]];
                kb1.u[j] = pack_bf(f.x, f.y);
            }
            s0 = wmma_bf16(qa[kk], kb0, s0);
            s1 = wmma_bf16(qa[kk], kb1, s1);
        }

        // online softmax per row (rows r+8*hb live in half-wave lane groups)
#pragma unroll
        for (int r = 0; r < 8; ++r) {
            int row = m0 + r + 8 * hb;
            float a0 = s0[r] * scale;
            float a1 = s1[r] * scale;
            if (t0 + l15 > row)      a0 = -3.0e38f;
            if (t0 + 16 + l15 > row) a1 = -3.0e38f;
            float mx = fmaxf(a0, a1);
#pragma unroll
            for (int off = 8; off >= 1; off >>= 1)
                mx = fmaxf(mx, __shfl_xor(mx, off, 32));
            float mnew  = fmaxf(mi[r], mx);
            float alpha = __expf(mi[r] - mnew);
            float p0 = __expf(a0 - mnew);
            float p1 = __expf(a1 - mnew);
            float ps = p0 + p1;
#pragma unroll
            for (int off = 8; off >= 1; off >>= 1)
                ps += __shfl_xor(ps, off, 32);
            li[r] = li[r] * alpha + ps;
            mi[r] = mnew;
#pragma unroll
            for (int cc = 0; cc < 8; ++cc) o[cc][r] *= alpha;
            sP[(r + 8 * hb) * 32 + l15]      = f2bf(p0);
            sP[(r + 8 * hb) * 32 + 16 + l15] = f2bf(p1);
        }
        __syncthreads();

        // P (16x32, C layout in LDS) -> A-fragment
        bf16frag pa;
#pragma unroll
        for (int j = 0; j < 8; ++j)
            pa.u[j] = *(const unsigned int*)&sP[l15 * 32 + kof[j]];

        // O += P @ V over 8 column chunks of head dim, V gathered from LDS
#pragma unroll
        for (int cc = 0; cc < 8; ++cc) {
            bf16frag vb;
#pragma unroll
            for (int j = 0; j < 8; ++j) {
                int kd = kof[j];
                float f0 = sV[kd * 128 + cc * 16 + l15];
                float f1 = sV[(kd + 1) * 128 + cc * 16 + l15];
                vb.u[j] = pack_bf(f0, f1);
            }
            o[cc] = wmma_bf16(pa, vb, o[cc]);
        }
        __syncthreads();
    }

#pragma unroll
    for (int cc = 0; cc < 8; ++cc)
#pragma unroll
        for (int r = 0; r < 8; ++r) {
            int row = m0 + r + 8 * hb;
            y[base + (size_t)row * W_ + cc * 16 + l15] = o[cc][r] / li[r];
        }
}

// ---------------- residual + LayerNorm (t2 optionally broadcast per batch) --
__global__ __launch_bounds__(256) void ln_res(
        float* __restrict__ x, const float* __restrict__ t2,
        const float* __restrict__ w, const float* __restrict__ b, int bcast) {
    __shared__ float red[256];
    int row = blockIdx.x;
    int tid = threadIdx.x;
    int n = row / T_;
    const float* t2r = t2 + (bcast ? (size_t)n * W_ : (size_t)row * W_);
    float* xr = x + (size_t)row * W_;

    float v0 = xr[tid] + t2r[tid];
    float v1 = xr[tid + 256] + t2r[tid + 256];
    red[tid] = v0 + v1;
    __syncthreads();
    for (int o = 128; o > 0; o >>= 1) {
        if (tid < o) red[tid] += red[tid + o];
        __syncthreads();
    }
    float mean = red[0] * (1.0f / W_);
    __syncthreads();
    float d0 = v0 - mean, d1 = v1 - mean;
    red[tid] = d0 * d0 + d1 * d1;
    __syncthreads();
    for (int o = 128; o > 0; o >>= 1) {
        if (tid < o) red[tid] += red[tid + o];
        __syncthreads();
    }
    float inv = rsqrtf(red[0] * (1.0f / W_) + 1e-5f);
    xr[tid]       = d0 * inv * w[tid]       + b[tid];
    xr[tid + 256] = d1 * inv * w[tid + 256] + b[tid + 256];
}

// ---------------- host orchestration ----------------
static inline void launch_gemm(const float* A, const float* B, const float* bias,
                               float* C, int M, int K, int Nn, int relu,
                               hipStream_t s) {
    dim3 grid(Nn / 128, M / 128);
    gemm_bf16<<<grid, 256, 0, s>>>(A, B, bias, C, K, Nn, relu);
}

extern "C" void kernel_launch(void* const* d_in, const int* in_sizes, int n_in,
                              void* d_out, int out_size, void* d_ws, size_t ws_size,
                              hipStream_t stream) {
    const float* features = (const float*)d_in[0];
    const int*   captions = (const int*)  d_in[1];
    const float* emb      = (const float*)d_in[2];
    const float* pe       = (const float*)d_in[3];
    const float* vis_w    = (const float*)d_in[4];
    const float* vis_b    = (const float*)d_in[5];
    const float* sa_wq = (const float*)d_in[6];   const float* sa_bq = (const float*)d_in[7];
    const float* sa_wk = (const float*)d_in[8];   const float* sa_bk = (const float*)d_in[9];
    const float* sa_wv = (const float*)d_in[10];  const float* sa_bv = (const float*)d_in[11];
    const float* sa_wo = (const float*)d_in[12];  const float* sa_bo = (const float*)d_in[13];
    const float* ca_wv = (const float*)d_in[18];  const float* ca_bv = (const float*)d_in[19];
    const float* ca_wo = (const float*)d_in[20];  const float* ca_bo = (const float*)d_in[21];
    const float* ff1_w = (const float*)d_in[22];  const float* ff1_b = (const float*)d_in[23];
    const float* ff2_w = (const float*)d_in[24];  const float* ff2_b = (const float*)d_in[25];
    const float* ln1_w = (const float*)d_in[26];  const float* ln1_b = (const float*)d_in[27];
    const float* ln2_w = (const float*)d_in[28];  const float* ln2_b = (const float*)d_in[29];
    const float* ln3_w = (const float*)d_in[30];  const float* ln3_b = (const float*)d_in[31];
    const float* out_w = (const float*)d_in[32];  const float* out_b = (const float*)d_in[33];

    float* ws = (float*)d_ws;
    const size_t XW = (size_t)NT_ * W_;
    float* x    = ws;                 // NT*W
    float* big  = ws + XW;            // 4*NT*W : q|k|v|y, aliased by FFN hidden
    float* qb   = big;
    float* kb   = big + XW;
    float* vb   = big + 2 * XW;
    float* yb   = big + 3 * XW;
    float* hbuf = big;                // NT*FF == 4*NT*W, reused after attention
    float* t2f  = ws + 5 * XW;        // NT*W
    float* mem  = ws + 6 * XW;        // N*W
    float* memv = mem + (size_t)N_ * W_;
    float* t2ca = memv + (size_t)N_ * W_;

    // x = emb[captions] + pe ; memory = features @ vis_w + vis_b
    embed_kernel<<<(NT_ * W_ + 255) / 256, 256, 0, stream>>>(captions, emb, pe, x);
    small_mm<<<(N_ * W_ + 255) / 256, 256, 0, stream>>>(features, vis_w, vis_b, mem, N_, D_, W_);

    for (int l = 0; l < L_; ++l) {
        const size_t ww = (size_t)l * W_ * W_, wb = (size_t)l * W_;
        // self-attention
        launch_gemm(x, sa_wq + ww, sa_bq + wb, qb, NT_, W_, W_, 0, stream);
        launch_gemm(x, sa_wk + ww, sa_bk + wb, kb, NT_, W_, W_, 0, stream);
        launch_gemm(x, sa_wv + ww, sa_bv + wb, vb, NT_, W_, W_, 0, stream);
        attn_causal<<<dim3(T_ / 16, H_, N_), 32, 0, stream>>>(qb, kb, vb, yb);
        launch_gemm(yb, sa_wo + ww, sa_bo + wb, t2f, NT_, W_, W_, 0, stream);
        ln_res<<<NT_, 256, 0, stream>>>(x, t2f, ln1_w + wb, ln1_b + wb, 0);
        // cross-attention over a single memory token collapses to (mem@Wv+bv)@Wo+bo
        small_mm<<<(N_ * W_ + 255) / 256, 256, 0, stream>>>(mem, ca_wv + ww, ca_bv + wb, memv, N_, W_, W_);
        small_mm<<<(N_ * W_ + 255) / 256, 256, 0, stream>>>(memv, ca_wo + ww, ca_bo + wb, t2ca, N_, W_, W_);
        ln_res<<<NT_, 256, 0, stream>>>(x, t2ca, ln2_w + wb, ln2_b + wb, 1);
        // FFN
        launch_gemm(x, ff1_w + (size_t)l * W_ * FF_, ff1_b + (size_t)l * FF_, hbuf, NT_, W_, FF_, 1, stream);
        launch_gemm(hbuf, ff2_w + (size_t)l * FF_ * W_, ff2_b + wb, t2f, NT_, FF_, W_, 0, stream);
        ln_res<<<NT_, 256, 0, stream>>>(x, t2f, ln3_w + wb, ln3_b + wb, 0);
    }
    // vocab projection -> d_out
    launch_gemm(x, out_w, out_b, (float*)d_out, NT_, W_, V_, 0, stream);
}
